// FlexibleIsolationEncodingLayer_68307159876092
// MI455X (gfx1250) — matrix operand order repaired and checked by reading, not compiled, not run on hardware
//
#include <hip/hip_runtime.h>

#define N_    65536
#define PSI_  1024
#define D_    256
#define BM    128
#define BN    128
#define GROUPS (N_ / 64)   // per-64-row column partial-sum groups

typedef __attribute__((ext_vector_type(16))) __bf16         v16bf;
typedef __attribute__((ext_vector_type(8)))  float          v8f;
typedef __attribute__((ext_vector_type(8)))  unsigned short u16x8;
typedef __attribute__((ext_vector_type(16))) unsigned short u16x16;

// ---- bf16 helpers (bit-exact RNE split of f32 into hi+lo bf16) ----
__device__ __forceinline__ unsigned short f2bf(float f) {
  unsigned u = __builtin_bit_cast(unsigned, f);
  return (unsigned short)((u + 0x7FFFu + ((u >> 16) & 1u)) >> 16);
}
__device__ __forceinline__ float bf2f(unsigned short h) {
  unsigned u = ((unsigned)h) << 16;
  return __builtin_bit_cast(float, u);
}
// Build a v16bf WMMA operand from two contiguous 16-byte chunks.
__device__ __forceinline__ v16bf ld16(const unsigned short* p0, const unsigned short* p1) {
  u16x8 a = *(const u16x8*)p0;
  u16x8 b = *(const u16x8*)p1;
  u16x16 c = __builtin_shufflevector(a, b, 0,1,2,3,4,5,6,7,8,9,10,11,12,13,14,15);
  return __builtin_bit_cast(v16bf, c);
}
__device__ __forceinline__ v8f wmma_bf16(v16bf a, v16bf b, v8f c) {
  return __builtin_amdgcn_wmma_f32_16x16x32_bf16(false, a, false, b, (short)0, c, false, false);
}

// ---- Kernel 1/2: split f32 rows (length 256) into bf16 hi/lo + row sq-norm ----
__global__ __launch_bounds__(256) void split_bf16_rownorm(
    const float* __restrict__ src, unsigned short* __restrict__ hi,
    unsigned short* __restrict__ lo, float* __restrict__ rn) {
  const int row = blockIdx.x, t = threadIdx.x;
  const size_t idx = (size_t)row * D_ + t;
  float f = src[idx];
  unsigned short h = f2bf(f);
  hi[idx] = h;
  lo[idx] = f2bf(f - bf2f(h));
  float p = f * f;
#pragma unroll
  for (int off = 16; off > 0; off >>= 1) p += __shfl_down(p, off, 32);
  __shared__ float red[8];
  if ((t & 31) == 0) red[t >> 5] = p;
  __syncthreads();
  if (t == 0) {
    float s = 0.f;
#pragma unroll
    for (int i = 0; i < 8; ++i) s += red[i];
    rn[row] = s;
  }
}

// ---- Kernel 3: WMMA GEMM (split bf16) + dist + exp epilogue + column partials ----
__global__ __launch_bounds__(256, 1) void gemm_dist_exp(
    const unsigned short* __restrict__ Xhi, const unsigned short* __restrict__ Xlo,
    const unsigned short* __restrict__ Shi, const unsigned short* __restrict__ Slo,
    const float* __restrict__ x2, const float* __restrict__ s2,
    const float* __restrict__ w, float* __restrict__ out,
    float* __restrict__ partial) {
  const int lane  = threadIdx.x & 31;
  const int wave  = threadIdx.x >> 5;
  const int wm    = wave & 1;        // 2 waves along M
  const int wn    = wave >> 1;       // 4 waves along N
  const int l15   = lane & 15;
  const int lhalf = lane >> 4;
  const int rowBase = blockIdx.y * BM + wm * 64;
  const int colBase = blockIdx.x * BN + wn * 32;

  v8f acc[4][2] = {};

  // A operand: lane l15 = row, 16 contiguous K at lhalf*16 (matches B)
  const unsigned short* aHi = Xhi + (size_t)(rowBase + l15) * D_ + lhalf * 16;
  const unsigned short* aLo = Xlo + (size_t)(rowBase + l15) * D_ + lhalf * 16;
  // B operand: lane l15 = column n = row of S, 16 contiguous K at lhalf*16
  const unsigned short* bHi = Shi + (size_t)(colBase + l15) * D_ + lhalf * 16;
  const unsigned short* bLo = Slo + (size_t)(colBase + l15) * D_ + lhalf * 16;

  for (int k0 = 0; k0 < D_; k0 += 32) {
    v16bf bh[2], bl[2];
#pragma unroll
    for (int ni = 0; ni < 2; ++ni) {
      const unsigned short* p = bHi + (size_t)ni * 16 * D_ + k0;
      bh[ni] = ld16(p, p + 8);
      const unsigned short* q = bLo + (size_t)ni * 16 * D_ + k0;
      bl[ni] = ld16(q, q + 8);
    }
#pragma unroll
    for (int mi = 0; mi < 4; ++mi) {
      const unsigned short* pa = aHi + (size_t)mi * 16 * D_ + k0;
      v16bf ah = ld16(pa, pa + 8);
      const unsigned short* qa = aLo + (size_t)mi * 16 * D_ + k0;
      v16bf al = ld16(qa, qa + 8);
#pragma unroll
      for (int ni = 0; ni < 2; ++ni) {
        acc[mi][ni] = wmma_bf16(ah, bh[ni], acc[mi][ni]);  // hi*hi
        acc[mi][ni] = wmma_bf16(ah, bl[ni], acc[mi][ni]);  // hi*lo
        acc[mi][ni] = wmma_bf16(al, bh[ni], acc[mi][ni]);  // lo*hi
      }
    }
  }

  // Epilogue: d2 = |x|^2 - 2g + |s|^2 ; E = exp(-sqrt(max(d2,0)) * w[col])
  const int col0 = colBase + l15;
  const int col1 = colBase + 16 + l15;
  const float s20 = s2[col0], s21 = s2[col1];
  const float w0 = w[col0], w1 = w[col1];
  float psum0 = 0.f, psum1 = 0.f;
#pragma unroll
  for (int mi = 0; mi < 4; ++mi) {
    const int rbase = rowBase + mi * 16 + lhalf * 8;  // C/D map: M = v + 8*lhalf
#pragma unroll
    for (int v = 0; v < 8; ++v) {
      const int row = rbase + v;
      const float xr = x2[row];
      {
        float g = acc[mi][0][v];
        float d2v = fmaxf(xr + s20 - 2.f * g, 0.f);
        float e = __expf(-sqrtf(d2v) * w0);
        out[(size_t)row * PSI_ + col0] = e;
        psum0 += e;
      }
      {
        float g = acc[mi][1][v];
        float d2v = fmaxf(xr + s21 - 2.f * g, 0.f);
        float e = __expf(-sqrtf(d2v) * w1);
        out[(size_t)row * PSI_ + col1] = e;
        psum1 += e;
      }
    }
  }
  // fold the two lane-halves (rows v and v+8) — deterministic, no atomics
  psum0 += __shfl_xor(psum0, 16, 32);
  psum1 += __shfl_xor(psum1, 16, 32);
  if (lane < 16) {
    const size_t group = (size_t)blockIdx.y * 2 + wm;  // 64 rows per group
    partial[group * PSI_ + col0] = psum0;
    partial[group * PSI_ + col1] = psum1;
  }
}

// ---- Kernel 4: colsum[j] = sum_g partial[g][j]  (deterministic tree) ----
__global__ __launch_bounds__(256) void reduce_cols(const float* __restrict__ partial,
                                                   float* __restrict__ colsum) {
  const int t = threadIdx.x;
  const int c = t & 63, q = t >> 6;
  const int col = blockIdx.x * 64 + c;
  float s = 0.f;
  for (int g = q * (GROUPS / 4); g < (q + 1) * (GROUPS / 4); ++g)
    s += partial[(size_t)g * PSI_ + col];
  __shared__ float sm[256];
  sm[t] = s;
  __syncthreads();
  if (t < 64)
    colsum[blockIdx.x * 64 + t] = sm[t] + sm[t + 64] + sm[t + 128] + sm[t + 192];
}

// ---- Kernel 5: out /= colsum (float4, column dim is contiguous) ----
__global__ __launch_bounds__(256) void normalize_k(float* __restrict__ out,
                                                   const float* __restrict__ colsum) {
  const size_t i = ((size_t)blockIdx.x * 256 + threadIdx.x) * 4;
  float4 v = *(float4*)(out + i);
  const float4 c = *(const float4*)(colsum + (i & (size_t)(PSI_ - 1)));
  v.x /= c.x; v.y /= c.y; v.z /= c.z; v.w /= c.w;
  *(float4*)(out + i) = v;
}

extern "C" void kernel_launch(void* const* d_in, const int* in_sizes, int n_in,
                              void* d_out, int out_size, void* d_ws, size_t ws_size,
                              hipStream_t stream) {
  (void)in_sizes; (void)n_in; (void)out_size; (void)ws_size;
  const float* X = (const float*)d_in[0];        // [N, D] f32
  const float* S = (const float*)d_in[1];        // [PSI, D] f32
  const float* W = (const float*)d_in[2];        // [1, PSI] f32
  float* out = (float*)d_out;                    // [N, PSI] f32

  char* w8 = (char*)d_ws;
  size_t off = 0;
  unsigned short* Xhi = (unsigned short*)(w8 + off); off += (size_t)N_ * D_ * 2;
  unsigned short* Xlo = (unsigned short*)(w8 + off); off += (size_t)N_ * D_ * 2;
  unsigned short* Shi = (unsigned short*)(w8 + off); off += (size_t)PSI_ * D_ * 2;
  unsigned short* Slo = (unsigned short*)(w8 + off); off += (size_t)PSI_ * D_ * 2;
  float* x2      = (float*)(w8 + off); off += (size_t)N_ * 4;
  float* s2      = (float*)(w8 + off); off += (size_t)PSI_ * 4;
  float* partial = (float*)(w8 + off); off += (size_t)GROUPS * PSI_ * 4;
  float* colsum  = (float*)(w8 + off); off += (size_t)PSI_ * 4;

  split_bf16_rownorm<<<N_, 256, 0, stream>>>(X, Xhi, Xlo, x2);
  split_bf16_rownorm<<<PSI_, 256, 0, stream>>>(S, Shi, Slo, s2);
  gemm_dist_exp<<<dim3(PSI_ / BN, N_ / BM), 256, 0, stream>>>(
      Xhi, Xlo, Shi, Slo, x2, s2, W, out, partial);
  reduce_cols<<<PSI_ / 64, 256, 0, stream>>>(partial, colsum);
  normalize_k<<<(N_ * (PSI_ / 4)) / 256, 256, 0, stream>>>(out, colsum);
}
